// PoolMixerBlock_71734543778460
// MI455X (gfx1250) — matrix operand order repaired and checked by reading, not compiled, not run on hardware
//
#include <hip/hip_runtime.h>
#include <hip/hip_bf16.h>

// dims from reference
#define B_ 8
#define C_ 512
#define S_ 64
#define T_ 128
#define D_ 256
#define P_ 8

typedef _Float16 v16h __attribute__((ext_vector_type(16)));
typedef _Float16 v8h  __attribute__((ext_vector_type(8)));
typedef _Float16 v4h  __attribute__((ext_vector_type(4)));
typedef float    v8f  __attribute__((ext_vector_type(8)));
typedef float    v4f  __attribute__((ext_vector_type(4)));
typedef int      v4i  __attribute__((ext_vector_type(4)));

#define LDW 40  // LDS row stride in halfs: 80 bytes, multiple of 16B for b128 loads

// ---------------- async copy (gfx1250 GLOBAL_LOAD_ASYNC_TO_LDS_B128) ----------------

__device__ __forceinline__ void async_copy16(const _Float16* g, _Float16* l) {
#if defined(__has_builtin) && __has_builtin(__builtin_amdgcn_global_load_async_to_lds_b128)
  __builtin_amdgcn_global_load_async_to_lds_b128(
      (__attribute__((address_space(1))) v4i*)(uintptr_t)g,
      (__attribute__((address_space(3))) v4i*)l, 0, 0);
#else
  *(v8h*)l = *(const v8h*)g;
#endif
}

__device__ __forceinline__ void wait_async() {
#if defined(__has_builtin) && __has_builtin(__builtin_amdgcn_s_wait_asynccnt)
  __builtin_amdgcn_s_wait_asynccnt(0);
#else
  asm volatile("s_wait_asynccnt 0x0" ::: "memory");
#endif
}

// ---------------- WMMA fragment helpers (CDNA5 16x16x32 f16 layouts) ----------------

__device__ __forceinline__ v16h frag_cat(v8h lo, v8h hi) {
  return __builtin_shufflevector(lo, hi, 0,1,2,3,4,5,6,7,8,9,10,11,12,13,14,15);
}

// A 16x32 (MxK): lane m = lane&15; lanes 0-15 hold K {0..7,16..23}, lanes 16-31 hold K {8..15,24..31}
__device__ __forceinline__ v16h load_a_frag(const _Float16* lds, int m0, int lane) {
  int m  = m0 + (lane & 15);
  int kb = (lane >> 4) * 8;
  v8h lo = *(const v8h*)(lds + m * LDW + kb);
  v8h hi = *(const v8h*)(lds + m * LDW + kb + 16);
  return frag_cat(lo, hi);
}

// B 32x16 (KxN), stored K-major per row n: lane n = lane&15; lanes 0-15 hold K {0..7,16..23}, lanes 16-31 hold K {8..15,24..31}
__device__ __forceinline__ v16h load_b_frag(const _Float16* lds, int n0, int lane) {
  int n  = n0 + (lane & 15);
  int kh = (lane >> 4) * 8;
  v8h lo = *(const v8h*)(lds + n * LDW + kh);
  v8h hi = *(const v8h*)(lds + n * LDW + kh + 16);
  return frag_cat(lo, hi);
}

// ---------------- K1: BN statistics (sum / sumsq per channel t) ----------------

__global__ void k_bnstats(const float* __restrict__ x, float* __restrict__ bnsum,
                          float* __restrict__ bnsumsq) {
  const int t = threadIdx.x;  // 128 channels
  float s = 0.f, sq = 0.f;
  const long nrows = (long)B_ * C_ * S_;
  for (long r = blockIdx.x; r < nrows; r += gridDim.x) {
    float v = x[r * T_ + t];
    s += v; sq += v * v;
  }
  atomicAdd(&bnsum[t], s);
  atomicAdd(&bnsumsq[t], sq);
}

// ---------------- K2: finalize BN affine + key norms ----------------

__global__ void k_bnfinalize(const float* __restrict__ bnsum, const float* __restrict__ bnsumsq,
                             const float* __restrict__ gamma, const float* __restrict__ beta,
                             const float* __restrict__ keys1, const float* __restrict__ keys2,
                             float* __restrict__ scale, float* __restrict__ shift,
                             float* __restrict__ k1n, float* __restrict__ k2n) {
  int t = threadIdx.x;  // 128
  const float invN = 1.0f / (float)((long)B_ * C_ * S_);
  float mean = bnsum[t] * invN;
  float var  = bnsumsq[t] * invN - mean * mean;
  float sc   = gamma[t] * rsqrtf(var + 1e-5f);
  scale[t] = sc;
  shift[t] = beta[t] - mean * sc;
  if (t < P_) {
    float s1 = 0.f;
    for (int i = 0; i < T_; ++i) { float v = keys1[t * T_ + i]; s1 += v * v; }
    k1n[t] = sqrtf(s1) + 1e-8f;
    float s2 = 0.f;
    for (int i = 0; i < D_; ++i) { float v = keys2[t * D_ + i]; s2 += v * v; }
    k2n[t] = sqrtf(s2) + 1e-8f;
  }
}

// ---------------- K3: pooled1 = mean_C(y) = affine(mean_C(x)) ----------------

__global__ void k_pool1(const float* __restrict__ x, const float* __restrict__ scale,
                        const float* __restrict__ shift, float* __restrict__ pooled1) {
  int bs = blockIdx.x; int b = bs / S_, s = bs % S_;
  int t = threadIdx.x;  // 128
  float acc = 0.f;
  const float* xp = x + (((long)b * C_) * S_ + s) * T_ + t;
  for (int c = 0; c < C_; ++c) acc += xp[(long)c * S_ * T_];
  pooled1[bs * T_ + t] = (acc * (1.0f / C_)) * scale[t] + shift[t];
}

// ---------------- routing: cosine sim vs P keys + softmax ----------------

template <int F>
__global__ void k_route(const float* __restrict__ pooled, float premul,
                        const float* __restrict__ keys, const float* __restrict__ knorm,
                        float* __restrict__ a_out) {
  __shared__ float red[F];
  __shared__ float sims[P_];
  int bs = blockIdx.x, t = threadIdx.x;
  float pv = pooled[bs * F + t] * premul;
  red[t] = pv * pv;
  __syncthreads();
  for (int off = F / 2; off > 0; off >>= 1) { if (t < off) red[t] += red[t + off]; __syncthreads(); }
  float pn = sqrtf(red[0]) + 1e-8f;
  for (int p = 0; p < P_; ++p) {
    __syncthreads();
    red[t] = pv * keys[p * F + t];
    __syncthreads();
    for (int off = F / 2; off > 0; off >>= 1) { if (t < off) red[t] += red[t + off]; __syncthreads(); }
    if (t == 0) sims[p] = red[0] / (pn * knorm[p]);
  }
  __syncthreads();
  if (t == 0) {
    float mx = -1e30f;
    for (int p = 0; p < P_; ++p) mx = fmaxf(mx, sims[p]);
    float e[P_]; float ssum = 0.f;
    for (int p = 0; p < P_; ++p) { e[p] = __expf(sims[p] - mx); ssum += e[p]; }
    for (int p = 0; p < P_; ++p) a_out[bs * P_ + p] = e[p] / ssum;
  }
}

// ---------------- K4/K7: mix weight bank with routing weights ----------------

__global__ void k_mixw1(const float* __restrict__ a1, const float* __restrict__ Wp1,
                        const float* __restrict__ bp1, float* __restrict__ w1_out,
                        float* __restrict__ b1) {
  int bs  = blockIdx.x;
  int idx = blockIdx.y * blockDim.x + threadIdx.x;  // < D_*T_  (d*T + t)
  float a[P_];
#pragma unroll
  for (int p = 0; p < P_; ++p) a[p] = a1[bs * P_ + p];
  float w = 0.f;
#pragma unroll
  for (int p = 0; p < P_; ++p) w += a[p] * Wp1[p * D_ * T_ + idx];
  w1_out[(long)bs * D_ * T_ + idx] = w;
  if ((idx & (T_ - 1)) == 0) {
    int d = idx / T_;
    float bb = 0.f;
#pragma unroll
    for (int p = 0; p < P_; ++p) bb += a[p] * bp1[p * D_ + d];
    b1[bs * D_ + d] = bb;
  }
}

__global__ void k_mixw2(const float* __restrict__ a2, const float* __restrict__ Wp2,
                        const float* __restrict__ bp2, float* __restrict__ w2_out,
                        float* __restrict__ b2) {
  int bs  = blockIdx.x;
  int idx = blockIdx.y * blockDim.x + threadIdx.x;  // < T_*D_  (t*D + d)
  float a[P_];
#pragma unroll
  for (int p = 0; p < P_; ++p) a[p] = a2[bs * P_ + p];
  float w = 0.f;
#pragma unroll
  for (int p = 0; p < P_; ++p) w += a[p] * Wp2[p * T_ * D_ + idx];
  w2_out[(long)bs * T_ * D_ + idx] = w;
  if ((idx & (D_ - 1)) == 0) {
    int t = idx / D_;
    float bb = 0.f;
#pragma unroll
    for (int p = 0; p < P_; ++p) bb += a[p] * bp2[p * T_ + t];
    b2[bs * T_ + t] = bb;
  }
}

// ---------------- K5: GEMM1 (C x T) @ (D x T)^T + b1, ReLU, fp16 y2, pooled2 sums ----------------

__global__ __launch_bounds__(256) void k_gemm1(
    const float* __restrict__ x, const float* __restrict__ scale, const float* __restrict__ shift,
    const float* __restrict__ w1, const float* __restrict__ b1,
    _Float16* __restrict__ y2, float* __restrict__ p2sum) {
  __shared__ __align__(16) _Float16 As[64 * LDW];
  __shared__ __align__(16) _Float16 Bs[256 * LDW];
  int bs = blockIdx.x; int b = bs / S_, s = bs % S_;
  int c0 = blockIdx.y * 64;
  int tid = threadIdx.x, lane = tid & 31, wave = tid >> 5;
  int m0 = (wave & 3) * 16;   // wave's 16 rows within the 64-row tile
  int ng = wave >> 2;         // wave's 128-col half of N=256
  const float* w1b = w1 + (long)bs * D_ * T_;
  const long xbase = (((long)b * C_) * S_ + s) * T_;
  v8f acc[8] = {};
  for (int k0 = 0; k0 < T_; k0 += 32) {
    __syncthreads();
    // A tile: 64x32 BN-normalized x, float4 loads + packed f16 LDS stores
#pragma unroll
    for (int i = 0; i < 2; ++i) {
      int e = tid + i * 256;
      int r = e >> 3, q = (e & 7) * 4;
      int t = k0 + q;
      v4f xv = *(const v4f*)(x + xbase + (long)(c0 + r) * (S_ * T_) + t);
      v4f sc = *(const v4f*)(scale + t);
      v4f sh = *(const v4f*)(shift + t);
      v4f yv = xv * sc + sh;
      *(v4h*)(As + r * LDW + q) = __builtin_convertvector(yv, v4h);
    }
    // B tile: 256x32 of w1 (f32 -> f16), float4 loads
#pragma unroll
    for (int i = 0; i < 8; ++i) {
      int e = tid + i * 256;
      int d = e >> 3, q = (e & 7) * 4;
      v4f wv = *(const v4f*)(w1b + d * T_ + k0 + q);
      *(v4h*)(Bs + d * LDW + q) = __builtin_convertvector(wv, v4h);
    }
    __syncthreads();
    v16h af = load_a_frag(As, m0, lane);
    v16h bf[8];
#pragma unroll
    for (int nt = 0; nt < 8; ++nt) bf[nt] = load_b_frag(Bs, ng * 128 + nt * 16, lane);
#pragma unroll
    for (int nt = 0; nt < 8; ++nt)
      acc[nt] = __builtin_amdgcn_wmma_f32_16x16x32_f16(false, af, false, bf[nt],
                                                       (short)0, acc[nt], false, false);
  }
  // epilogue: +b1, ReLU, store fp16 y2, accumulate column sums for pooled2
  int mrow = c0 + m0 + ((lane >> 4) ? 8 : 0);
#pragma unroll
  for (int nt = 0; nt < 8; ++nt) {
    int n = ng * 128 + nt * 16 + (lane & 15);
    float bv = b1[bs * D_ + n];
    float colsum = 0.f;
#pragma unroll
    for (int r = 0; r < 8; ++r) {
      float v = acc[nt][r] + bv;
      v = fmaxf(v, 0.f);
      colsum += v;
      int c = mrow + r;
      y2[(((long)b * C_ + c) * S_ + s) * D_ + n] = (_Float16)v;
    }
    atomicAdd(&p2sum[bs * D_ + n], colsum);
  }
}

// ---------------- K8: GEMM2 (C x D) @ (T x D)^T + b2 + residual ----------------

__global__ __launch_bounds__(256) void k_gemm2(
    const float* __restrict__ x, const _Float16* __restrict__ y2,
    const float* __restrict__ w2, const float* __restrict__ b2,
    float* __restrict__ out0) {
  __shared__ __align__(16) _Float16 As[64 * LDW];
  __shared__ __align__(16) _Float16 Bs[128 * LDW];
  int bs = blockIdx.x; int b = bs / S_, s = bs % S_;
  int c0 = blockIdx.y * 64;
  int tid = threadIdx.x, lane = tid & 31, wave = tid >> 5;
  int m0 = (wave & 3) * 16;
  int ng = wave >> 2;  // 64-col half of N=128
  const float* w2b = w2 + (long)bs * T_ * D_;
  const long ybase = (((long)b * C_) * S_ + s) * D_;
  v8f acc[4] = {};
  for (int k0 = 0; k0 < D_; k0 += 32) {
    __syncthreads();
    // A tile: 64x32 fp16 y2 -> LDS via async DMA (16B per thread, ASYNCcnt-tracked)
    {
      int r = tid >> 2, ch = (tid & 3) * 8;
      const _Float16* g = y2 + ybase + (long)(c0 + r) * (S_ * D_) + k0 + ch;
      async_copy16(g, As + r * LDW + ch);
    }
    // B tile: 128x32 of w2 (f32 -> f16), float4 loads
#pragma unroll
    for (int i = 0; i < 4; ++i) {
      int e = tid + i * 256;
      int t = e >> 3, q = (e & 7) * 4;
      v4f wv = *(const v4f*)(w2b + t * D_ + k0 + q);
      *(v4h*)(Bs + t * LDW + q) = __builtin_convertvector(wv, v4h);
    }
    wait_async();
    __syncthreads();
    v16h af = load_a_frag(As, m0, lane);
    v16h bf[4];
#pragma unroll
    for (int nt = 0; nt < 4; ++nt) bf[nt] = load_b_frag(Bs, ng * 64 + nt * 16, lane);
#pragma unroll
    for (int nt = 0; nt < 4; ++nt)
      acc[nt] = __builtin_amdgcn_wmma_f32_16x16x32_f16(false, af, false, bf[nt],
                                                       (short)0, acc[nt], false, false);
  }
  int mrow = c0 + m0 + ((lane >> 4) ? 8 : 0);
#pragma unroll
  for (int nt = 0; nt < 4; ++nt) {
    int n = ng * 64 + nt * 16 + (lane & 15);
    float bv = b2[bs * T_ + n];
#pragma unroll
    for (int r = 0; r < 8; ++r) {
      int c = mrow + r;
      long idx = (((long)b * C_ + c) * S_ + s) * T_ + n;
      out0[idx] = x[idx] + bv + acc[nt][r];
    }
  }
}

// ---------------- host: pipeline ----------------

extern "C" void kernel_launch(void* const* d_in, const int* in_sizes, int n_in,
                              void* d_out, int out_size, void* d_ws, size_t ws_size,
                              hipStream_t stream) {
  const float* x     = (const float*)d_in[0];
  const float* gamma = (const float*)d_in[1];
  const float* beta  = (const float*)d_in[2];
  const float* keys1 = (const float*)d_in[3];
  const float* Wp1   = (const float*)d_in[4];
  const float* bp1   = (const float*)d_in[5];
  const float* keys2 = (const float*)d_in[6];
  const float* Wp2   = (const float*)d_in[7];
  const float* bp2   = (const float*)d_in[8];

  float* out0 = (float*)d_out;                       // (B,C,S,T)
  float* w1   = out0 + (long)B_ * C_ * S_ * T_;      // (B,S,D,T)
  float* w2   = w1 + (long)B_ * S_ * D_ * T_;        // (B,S,T,D)

  float* f = (float*)d_ws;
  float* bnsum   = f;                 // 128
  float* bnsumsq = f + 128;           // 128
  float* scale   = f + 256;           // 128
  float* shift   = f + 384;           // 128
  float* a1      = f + 512;           // 4096
  float* a2      = f + 4608;          // 4096
  float* b1      = f + 8704;          // 131072
  float* b2      = f + 139776;        // 65536
  float* p2sum   = f + 205312;        // 131072
  float* pooled1 = f + 336384;        // 65536
  float* k1n     = f + 401920;        // 8
  float* k2n     = f + 401928;        // 8
  _Float16* y2bf = (_Float16*)(f + 401936);  // B*C*S*D halfs (134 MB), 16B-aligned offset

  // zero the atomic accumulation regions (graph-capturable)
  (void)hipMemsetAsync(bnsum, 0, 2 * T_ * sizeof(float), stream);
  (void)hipMemsetAsync(p2sum, 0, (size_t)B_ * S_ * D_ * sizeof(float), stream);

  k_bnstats<<<256, 128, 0, stream>>>(x, bnsum, bnsumsq);
  k_bnfinalize<<<1, 128, 0, stream>>>(bnsum, bnsumsq, gamma, beta, keys1, keys2,
                                      scale, shift, k1n, k2n);
  k_pool1<<<B_ * S_, T_, 0, stream>>>(x, scale, shift, pooled1);
  k_route<T_><<<B_ * S_, T_, 0, stream>>>(pooled1, 1.0f, keys1, k1n, a1);
  dim3 g4(B_ * S_, (D_ * T_) / 256);
  k_mixw1<<<g4, 256, 0, stream>>>(a1, Wp1, bp1, w1, b1);
  dim3 g5(B_ * S_, C_ / 64);
  k_gemm1<<<g5, 256, 0, stream>>>(x, scale, shift, w1, b1, y2bf, p2sum);
  k_route<D_><<<B_ * S_, D_, 0, stream>>>(p2sum, 1.0f / (float)C_, keys2, k2n, a2);
  dim3 g7(B_ * S_, (T_ * D_) / 256);
  k_mixw2<<<g7, 256, 0, stream>>>(a2, Wp2, bp2, w2, b2);
  k_gemm2<<<g5, 256, 0, stream>>>(x, y2bf, w2, b2, out0);
}
